// Attention_1194000909013
// MI455X (gfx1250) — compile-verified, matrix operand order checked by reading, not run on hardware
//
#include <hip/hip_runtime.h>

// ---------------- types ----------------
typedef __attribute__((ext_vector_type(16))) __bf16 v16bf;
typedef __attribute__((ext_vector_type(8)))  float  v8f;
typedef __attribute__((ext_vector_type(4)))  float  v4f;

#define CDIM 512
#define DDIM 64
#define NSP  4096
#define BATCHB 8
#define BNROWS (BATCHB * NSP)   // 32768
#define LOG2E 1.4426950408889634f

static __device__ __forceinline__ __bf16 f2bf(float f) {
    unsigned u = __builtin_bit_cast(unsigned, f);
    unsigned r = (u + 0x7FFFu + ((u >> 16) & 1u)) >> 16;
    return __builtin_bit_cast(__bf16, (unsigned short)r);
}
// truncating f32->bf16 store (folds to *_store_d16_hi_b16)
static __device__ __forceinline__ void st_bf16_trunc(__bf16* p, float v) {
    unsigned u = __builtin_bit_cast(unsigned, v);
    *(unsigned short*)p = (unsigned short)(u >> 16);
}

struct U32x2 { uint4 a, b; };

// 16 contiguous bf16 (32 bytes) -> v16bf
static __device__ __forceinline__ v16bf ldb_contig(const __bf16* p) {
    U32x2 u;
    u.a = *(const uint4*)(p);
    u.b = *(const uint4*)(p + 8);
    return __builtin_bit_cast(v16bf, u);
}
// split layout: elems 0..7 at p, elems 8..15 at p+16 (A/B natural striping)
static __device__ __forceinline__ v16bf ldb_split(const __bf16* p) {
    U32x2 u;
    u.a = *(const uint4*)(p);
    u.b = *(const uint4*)(p + 16);
    return __builtin_bit_cast(v16bf, u);
}

// =====================================================================
// Kernel 0: pre-swizzle a K x N f32 weight matrix into bf16 B-operand
// layout: [kc = K/32][t = N/16][lane 0..31][16 contiguous elems].
// =====================================================================
__global__ __launch_bounds__(256) void wprep_kernel(
    const float* __restrict__ W, __bf16* __restrict__ out, int K, int N)
{
    const int idx = blockIdx.x * blockDim.x + threadIdx.x;
    if (idx >= K * N) return;
    const int e    = idx & 15;
    const int lane = (idx >> 4) & 31;
    const int rest = idx >> 9;
    const int nt   = N >> 4;
    const int t    = rest % nt;
    const int kc   = rest / nt;
    const int k = kc * 32 + (e >> 3) * 16 + (lane >> 4) * 8 + (e & 7);
    const int n = t * 16 + (lane & 15);
    out[idx] = f2bf(W[(size_t)k * N + n]);
}

// =====================================================================
// Kernel 1: f = (X@Wf+bf)*log2(e), g = X@Wg+bg, h = X@Wh+bh
// f,g row-major bf16 [32768,64]; h transposed per batch: hT[b][d][n].
// (log2(e) folded into Q so attention can use native base-2 exp.)
// =====================================================================
__global__ __launch_bounds__(256) void proj_kernel(
    const float* __restrict__ x,
    const __bf16* __restrict__ WswAll,
    const float* __restrict__ bfv, const float* __restrict__ bgv,
    const float* __restrict__ bhv,
    __bf16* __restrict__ fOut, __bf16* __restrict__ gOut,
    __bf16* __restrict__ hT)
{
    const int lane = threadIdx.x & 31;
    const int wid  = blockIdx.x * (blockDim.x >> 5) + (threadIdx.x >> 5);
    const int row0 = wid * 16;
    const int half = lane >> 4;
    const int l16  = lane & 15;

    v8f zero = {};
    v8f acc[3][4];
    #pragma unroll
    for (int p = 0; p < 3; ++p)
        #pragma unroll
        for (int t = 0; t < 4; ++t) acc[p][t] = zero;

    for (int k0 = 0; k0 < CDIM; k0 += 32) {
        v16bf a;
        const float* ax = x + (size_t)(row0 + l16) * CDIM + k0 + half * 8;
        #pragma unroll
        for (int hh = 0; hh < 2; ++hh) {
            v4f t0 = *(const v4f*)(ax + hh * 16);
            v4f t1 = *(const v4f*)(ax + hh * 16 + 4);
            #pragma unroll
            for (int j = 0; j < 4; ++j) {
                a[hh * 8 + j]     = f2bf(t0[j]);
                a[hh * 8 + 4 + j] = f2bf(t1[j]);
            }
        }
        const int kc = k0 >> 5;
        #pragma unroll
        for (int p = 0; p < 3; ++p) {
            const __bf16* Wsw = WswAll + (size_t)p * (CDIM * DDIM);
            #pragma unroll
            for (int t = 0; t < 4; ++t) {
                v16bf bmat = ldb_contig(Wsw + (((kc * 4 + t) * 32) + lane) * 16);
                acc[p][t] = __builtin_amdgcn_wmma_f32_16x16x32_bf16(
                    false, a, false, bmat, (short)0, acc[p][t], false, false);
            }
        }
    }

    #pragma unroll
    for (int t = 0; t < 4; ++t) {
        const int col = t * 16 + l16;
        const float b0 = bfv[col], b1 = bgv[col], b2 = bhv[col];
        #pragma unroll
        for (int r = 0; r < 8; ++r) {
            const int row = row0 + r + half * 8;
            fOut[(size_t)row * DDIM + col] = f2bf((acc[0][t][r] + b0) * LOG2E);
            gOut[(size_t)row * DDIM + col] = f2bf(acc[1][t][r] + b1);
            const int bidx = row >> 12, nidx = row & (NSP - 1);
            hT[((size_t)(bidx * DDIM + col) << 12) + nidx] = f2bf(acc[2][t][r] + b2);
        }
    }
}

// =====================================================================
// Kernel 2: fused flash attention per batch (64-key chunks).
// One wave owns 16 query rows. Per iteration:
//   4 S tiles (8 wmma), base-2 online softmax (shfl max tree only),
//   P truncated to bf16 in LDS, ctx += P@V (8 wmma), rowsum += P@1
//   (2 wmma)  -- no shuffle-based row sums needed.
// =====================================================================
__global__ __launch_bounds__(128) void attn_kernel(
    const __bf16* __restrict__ fQ, const __bf16* __restrict__ gK,
    const __bf16* __restrict__ hT, __bf16* __restrict__ ctx)
{
    __shared__ __bf16 lds[4][16 * 64];
    const int lane = threadIdx.x & 31;
    const int w    = threadIdx.x >> 5;
    const int wid  = blockIdx.x * 4 + w;
    const int row0 = wid * 16;
    const int b    = row0 >> 12;
    const int half = lane >> 4;
    const int l16  = lane & 15;

    const __bf16* gB = gK + ((size_t)b << 12) * DDIM;   // g rows of this batch
    const __bf16* vB = hT + (size_t)b * DDIM * NSP;     // hT[64][4096] of batch

    // Q as A-operand, 2 d-chunks of K=32  (already scaled by log2(e))
    const __bf16* qp = fQ + (size_t)(row0 + l16) * DDIM + half * 8;
    v16bf qa[2];
    qa[0] = ldb_split(qp);
    qa[1] = ldb_split(qp + 32);

    // B operand of all-ones (bf16 1.0 = 0x3F80) for row-sum WMMA
    v16bf ones;
    {
        U32x2 u;
        u.a = make_uint4(0x3F803F80u, 0x3F803F80u, 0x3F803F80u, 0x3F803F80u);
        u.b = u.a;
        ones = __builtin_bit_cast(v16bf, u);
    }

    v8f zero = {};
    v8f o[4];
    #pragma unroll
    for (int t = 0; t < 4; ++t) o[t] = zero;
    v8f osum = zero;
    float mrow[8];
    #pragma unroll
    for (int r = 0; r < 8; ++r) mrow[r] = -1e30f;

    for (int m0 = 0; m0 < NSP; m0 += 64) {
        // ---- 4 S tiles: cols [m0+16*kt, m0+16*kt+16) ----
        v8f s[4];
        #pragma unroll
        for (int kt = 0; kt < 4; ++kt) {
            const __bf16* kp = gB + (size_t)(m0 + kt * 16 + l16) * DDIM + half * 8;
            v8f sa = zero;
            #pragma unroll
            for (int dc = 0; dc < 2; ++dc) {
                v16bf kb = ldb_split(kp + dc * 32);
                sa = __builtin_amdgcn_wmma_f32_16x16x32_bf16(
                    false, qa[dc], false, kb, (short)0, sa, false, false);
            }
            s[kt] = sa;
        }

        // ---- online softmax (base-2), per row r (row = r + half*8) ----
        #pragma unroll
        for (int r = 0; r < 8; ++r) {
            float v = fmaxf(fmaxf(s[0][r], s[1][r]), fmaxf(s[2][r], s[3][r]));
            #pragma unroll
            for (int msk = 8; msk >= 1; msk >>= 1)
                v = fmaxf(v, __shfl_xor(v, msk, 32));
            const float mn = fmaxf(mrow[r], v);
            const float alpha = exp2f(mrow[r] - mn);
            mrow[r] = mn;
            #pragma unroll
            for (int kt = 0; kt < 4; ++kt)
                s[kt][r] = exp2f(s[kt][r] - mn);
            #pragma unroll
            for (int t = 0; t < 4; ++t) o[t][r] *= alpha;
            osum[r] *= alpha;
        }

        // ---- stage P (16x64) into LDS, truncating f32->bf16 ----
        __bf16* L = lds[w];
        #pragma unroll
        for (int r = 0; r < 8; ++r) {
            const int n = r + half * 8;
            #pragma unroll
            for (int kt = 0; kt < 4; ++kt)
                st_bf16_trunc(&L[n * 64 + kt * 16 + l16], s[kt][r]);
        }
        __builtin_amdgcn_wave_barrier();
        v16bf pa[2];
        pa[0] = ldb_split(L + l16 * 64 + half * 8);
        pa[1] = ldb_split(L + l16 * 64 + 32 + half * 8);
        __builtin_amdgcn_wave_barrier();

        // ---- ctx += P @ V ; rowsum += P @ 1 ----
        #pragma unroll
        for (int c = 0; c < 2; ++c) {
            #pragma unroll
            for (int t = 0; t < 4; ++t) {
                const int d = t * 16 + l16;
                v16bf vb = ldb_split(vB + (size_t)d * NSP + m0 + c * 32 + half * 8);
                o[t] = __builtin_amdgcn_wmma_f32_16x16x32_bf16(
                    false, pa[c], false, vb, (short)0, o[t], false, false);
            }
            osum = __builtin_amdgcn_wmma_f32_16x16x32_bf16(
                false, pa[c], false, ones, (short)0, osum, false, false);
        }
    }

    // ---- finalize: divide by row sums, store ctx (bf16 row-major) ----
    #pragma unroll
    for (int r = 0; r < 8; ++r) {
        const float inv = 1.0f / osum[r];
        const int row = row0 + r + half * 8;
        #pragma unroll
        for (int t = 0; t < 4; ++t) {
            const int col = t * 16 + l16;
            ctx[(size_t)row * DDIM + col] = f2bf(o[t][r] * inv);
        }
    }
}

// =====================================================================
// Kernel 3: out = gamma*(ctx @ Wv) + bv + x   (f32 output, residual)
// Wv pre-swizzled: [kc=2][t=32][lane][16] bf16.
// =====================================================================
__global__ __launch_bounds__(256) void outproj_kernel(
    const __bf16* __restrict__ ctx, const __bf16* __restrict__ WvSw,
    const float* __restrict__ bvv, const float* __restrict__ x,
    const float* __restrict__ gammap, float* __restrict__ out)
{
    const int lane = threadIdx.x & 31;
    const int wid  = blockIdx.x * (blockDim.x >> 5) + (threadIdx.x >> 5);
    const int rowt = wid >> 3;
    const int cg   = wid & 7;
    const int row0 = rowt * 16;
    const int half = lane >> 4;
    const int l16  = lane & 15;
    const float gamma = gammap[0];

    const __bf16* ap = ctx + (size_t)(row0 + l16) * DDIM + half * 8;
    v16bf a[2];
    a[0] = ldb_split(ap);
    a[1] = ldb_split(ap + 32);

    v8f zero = {};
    v8f acc[4];
    #pragma unroll
    for (int t = 0; t < 4; ++t) acc[t] = zero;

    #pragma unroll
    for (int dc = 0; dc < 2; ++dc) {
        #pragma unroll
        for (int t = 0; t < 4; ++t) {
            const int tglob = cg * 4 + t;
            v16bf bmat = ldb_contig(WvSw + (((dc * 32 + tglob) * 32) + lane) * 16);
            acc[t] = __builtin_amdgcn_wmma_f32_16x16x32_bf16(
                false, a[dc], false, bmat, (short)0, acc[t], false, false);
        }
    }

    #pragma unroll
    for (int t = 0; t < 4; ++t) {
        const int c = cg * 64 + t * 16 + l16;
        const float bias = bvv[c];
        #pragma unroll
        for (int r = 0; r < 8; ++r) {
            const size_t row = (size_t)(row0 + r + half * 8);
            out[row * CDIM + c] = gamma * acc[t][r] + bias + x[row * CDIM + c];
        }
    }
}

// =====================================================================
extern "C" void kernel_launch(void* const* d_in, const int* in_sizes, int n_in,
                              void* d_out, int out_size, void* d_ws, size_t ws_size,
                              hipStream_t stream) {
    (void)in_sizes; (void)n_in; (void)out_size; (void)ws_size;
    const float* x  = (const float*)d_in[0];
    const float* Wf = (const float*)d_in[1];
    const float* bf = (const float*)d_in[2];
    const float* Wg = (const float*)d_in[3];
    const float* bg = (const float*)d_in[4];
    const float* Wh = (const float*)d_in[5];
    const float* bh = (const float*)d_in[6];
    const float* Wv = (const float*)d_in[7];
    const float* bv = (const float*)d_in[8];
    const float* gm = (const float*)d_in[9];
    float* out = (float*)d_out;

    // Workspace layout (16 MB activations + 256 KB swizzled weights)
    const size_t slab = (size_t)BNROWS * DDIM * sizeof(__bf16);  // 4 MB
    const size_t wsl  = (size_t)CDIM * DDIM * sizeof(__bf16);    // 64 KB
    char* ws = (char*)d_ws;
    __bf16* fO   = (__bf16*)(ws);
    __bf16* gO   = (__bf16*)(ws + slab);
    __bf16* hT   = (__bf16*)(ws + 2 * slab);
    __bf16* ctx  = (__bf16*)(ws + 3 * slab);
    __bf16* WfSw = (__bf16*)(ws + 4 * slab);
    __bf16* WgSw = (__bf16*)(ws + 4 * slab + wsl);
    __bf16* WhSw = (__bf16*)(ws + 4 * slab + 2 * wsl);
    __bf16* WvSw = (__bf16*)(ws + 4 * slab + 3 * wsl);

    wprep_kernel<<<(CDIM * DDIM) / 256, 256, 0, stream>>>(Wf, WfSw, CDIM, DDIM);
    wprep_kernel<<<(CDIM * DDIM) / 256, 256, 0, stream>>>(Wg, WgSw, CDIM, DDIM);
    wprep_kernel<<<(CDIM * DDIM) / 256, 256, 0, stream>>>(Wh, WhSw, CDIM, DDIM);
    wprep_kernel<<<(DDIM * CDIM) / 256, 256, 0, stream>>>(Wv, WvSw, DDIM, CDIM);

    proj_kernel<<<BNROWS / (16 * 8), 256, 0, stream>>>(
        x, WfSw, bf, bg, bh, fO, gO, hT);
    attn_kernel<<<BNROWS / (16 * 4), 128, 0, stream>>>(fO, gO, hT, ctx);
    outproj_kernel<<<(BNROWS / 16) * 8 / 8, 256, 0, stream>>>(
        ctx, WvSw, bv, x, gm, out);
}
